// coarsesGenerator_82162724373118
// MI455X (gfx1250) — matrix operand order repaired and checked by reading, not compiled, not run on hardware
//
#include <hip/hip_runtime.h>
#include <hip/hip_bf16.h>

// ---------------------------------------------------------------------------
// CDNA5 (gfx1250) coarse-generator pipeline.
// All dense layers: v_wmma_f32_16x16x32_bf16 (bf16 in, f32 accumulate).
// GEMMs are register-blocked 64x16 per wave (B fragment loaded once per
// k-chunk, reused by 4 WMMAs). EdgeConv fused per point with LDS bf16 tiles.
// __launch_bounds__(32,1): one wave per workgroup -> full VGPR budget, no spills.
// ---------------------------------------------------------------------------

typedef __attribute__((ext_vector_type(16))) __bf16 v16bf;
typedef __attribute__((ext_vector_type(8)))  __bf16 v8bf;
typedef __attribute__((ext_vector_type(8)))  float  v8f;
typedef __attribute__((ext_vector_type(8)))  float  float8v;
typedef __attribute__((ext_vector_type(4)))  float  float4v;

#define B_    4
#define N_    4096
#define BN_   16384          // B*N
#define M2_   65536          // B*N*RATE
#define KNN_K 24

// grid for RATE=4 -> 2x2 meshgrid of linspace(-0.2,0.2,2)
__constant__ float c_grid[8] = {-0.2f, -0.2f, -0.2f, 0.2f, 0.2f, -0.2f, 0.2f, 0.2f};

__device__ __forceinline__ __bf16 f2bf(float f) { return (__bf16)f; }
__device__ __forceinline__ float  bf2f(__bf16 h) { return (float)h; }

__device__ __forceinline__ v8bf cvt8(float4v a, float4v b) {
  float8v f = __builtin_shufflevector(a, b, 0, 1, 2, 3, 4, 5, 6, 7);
  return __builtin_convertvector(f, v8bf);   // lowers to v_cvt_pk_bf16_f32
}
__device__ __forceinline__ v16bf pack16(v8bf lo, v8bf hi) {
  return __builtin_shufflevector(lo, hi, 0, 1, 2, 3, 4, 5, 6, 7, 8, 9, 10, 11, 12, 13, 14, 15);
}

// ---- WMMA fragment loaders (layouts per CDNA5 ISA §7.12.2) -----
// A 16x32 bf16: lanes 0-15 row M=lane, elems 0..7 = K kb..kb+7, 8..15 = K kb+16..kb+23
//               with kb = 0 (lanes<16) / 8 (lanes>=16).
// B 32x16 bf16: col N = lane%16, same K order as A (elems 0..7 = kb..kb+7, 8..15 = kb+16..kb+23).
// C/D f32 x8  : elem v -> (M = v + 8*(lane>=16), N = lane%16).

// from a per-lane row pointer (already offset by row*lda + lane-half k offset)
__device__ __forceinline__ v16bf load_a_vec(const float* p) {
  float4v x0 = *(const float4v*)(p);
  float4v x1 = *(const float4v*)(p + 4);
  float4v x2 = *(const float4v*)(p + 16);
  float4v x3 = *(const float4v*)(p + 20);
  return pack16(cvt8(x0, x1), cvt8(x2, x3));
}

// tail: clamp addresses only; garbage beyond K is annihilated by zeroed B rows.
__device__ __forceinline__ v16bf load_a_tail(const float* __restrict__ A, int lda, int K,
                                             int m0, int k0, int lane) {
  int row = m0 + (lane & 15);
  int kb  = k0 + ((lane & 16) ? 8 : 0);
  const float* Ar = A + (size_t)row * lda;
  v16bf a;
#pragma unroll
  for (int j = 0; j < 8; ++j) { int k = kb + j;      a[j]     = f2bf(Ar[k < K ? k : K - 1]); }
#pragma unroll
  for (int j = 0; j < 8; ++j) { int k = kb + 16 + j; a[8 + j] = f2bf(Ar[k < K ? k : K - 1]); }
  return a;
}

// n pre-clamped (< Nc) by caller. Independent indexed addresses -> one clause.
__device__ __forceinline__ v16bf load_b_fast(const float* __restrict__ W, int Nc,
                                             int k0, int n, int lane) {
  int kb = k0 + ((lane & 16) ? 8 : 0);
  unsigned base = (unsigned)(kb * Nc + n);
  v16bf b;
#pragma unroll
  for (int j = 0; j < 16; ++j) b[j] = f2bf(W[base + (unsigned)((j + (j & 8)) * Nc)]);
  return b;
}

__device__ __forceinline__ v16bf load_b_tail(const float* __restrict__ W, int K, int Nc,
                                             int k0, int n, int lane) {
  int kb = k0 + ((lane & 16) ? 8 : 0);
  v16bf b;
#pragma unroll
  for (int j = 0; j < 16; ++j) {
    int k = kb + j + (j & 8);
    int kcl = k < K ? k : K - 1;
    float wv = W[(unsigned)(kcl * Nc + n)];
    b[j] = f2bf(k < K ? wv : 0.0f);
  }
  return b;
}

// A tile from LDS bf16 matrix: two aligned ds_load_b128.
template <int LD>
__device__ __forceinline__ v16bf load_a_lds(const __bf16 (*S)[LD], int m0, int k0, int lane) {
  int row = m0 + (lane & 15);
  int kb  = k0 + ((lane & 16) ? 8 : 0);
  const __bf16* p = &S[row][kb];
  v8bf lo = *(const v8bf*)p;
  v8bf hi = *(const v8bf*)(p + 16);
  return pack16(lo, hi);
}

#define WMMA_BF16(a, b, c) \
  __builtin_amdgcn_wmma_f32_16x16x32_bf16(false, (a), false, (b), (short)0, (c), false, false)

// ---------------------------------------------------------------------------
// KNN: each thread owns one query point; candidate points tiled through LDS.
// ---------------------------------------------------------------------------
__global__ __launch_bounds__(128, 1)
void knn_kernel(const float* __restrict__ P, int* __restrict__ IDX) {
  int t = blockIdx.x * blockDim.x + threadIdx.x;   // 0..BN_-1
  int b = t >> 12, n = t & (N_ - 1);
  const float* Pb = P + (size_t)b * N_ * 3;
  float px = P[(size_t)t * 3 + 0];
  float py = P[(size_t)t * 3 + 1];
  float pz = P[(size_t)t * 3 + 2];

  float bd[KNN_K]; int bi[KNN_K];
#pragma unroll
  for (int i = 0; i < KNN_K; ++i) { bd[i] = 3.4e38f; bi[i] = 0; }

  __shared__ float sp[128 * 3];
  for (int tile = 0; tile < N_; tile += 128) {
    __syncthreads();
    int m = tile + threadIdx.x;
    sp[threadIdx.x * 3 + 0] = Pb[(size_t)m * 3 + 0];
    sp[threadIdx.x * 3 + 1] = Pb[(size_t)m * 3 + 1];
    sp[threadIdx.x * 3 + 2] = Pb[(size_t)m * 3 + 2];
    __syncthreads();
    for (int j = 0; j < 128; ++j) {
      int m2 = tile + j;
      float dx = px - sp[j * 3 + 0];
      float dy = py - sp[j * 3 + 1];
      float dz = pz - sp[j * 3 + 2];
      float d2 = dx * dx + dy * dy + dz * dz;
      if (m2 != n && d2 < bd[KNN_K - 1]) {
        bd[KNN_K - 1] = d2; bi[KNN_K - 1] = m2;
#pragma unroll
        for (int i = KNN_K - 1; i > 0; --i) {
          if (bd[i] < bd[i - 1]) {
            float tf = bd[i]; bd[i] = bd[i - 1]; bd[i - 1] = tf;
            int   ti = bi[i]; bi[i] = bi[i - 1]; bi[i - 1] = ti;
          }
        }
      }
    }
  }
#pragma unroll
  for (int i = 0; i < KNN_K; ++i) IDX[(size_t)t * KNN_K + i] = bi[i];
}

// ---------------------------------------------------------------------------
// Generic GEMM: C[:, coff:coff+Nc] = act(A[M x K] @ W[K x Nc] + bias)
// One wave per 64x16 tile (4 m-subtiles share each B fragment). M % 64 == 0.
// ---------------------------------------------------------------------------
__global__ __launch_bounds__(32, 1)
void gemm_bias_act(const float* __restrict__ A, int lda, int K,
                   const float* __restrict__ W, int Nc,
                   const float* __restrict__ bias,
                   float* __restrict__ C, int ldc, int coff, int relu) {
  int lane = threadIdx.x;
  int m0 = blockIdx.x * 64;
  int n0 = blockIdx.y * 16;
  int n = n0 + (lane & 15);
  int ncl = n < Nc ? n : Nc - 1;
  __builtin_prefetch(W, 0, 3);

  // per-lane A tile base pointers (row + lane-half k offset), stepped by 32 floats
  int khalf = (lane & 16) ? 8 : 0;
  const float* ap0 = A + (size_t)(m0 +  0 + (lane & 15)) * lda + khalf;
  const float* ap1 = A + (size_t)(m0 + 16 + (lane & 15)) * lda + khalf;
  const float* ap2 = A + (size_t)(m0 + 32 + (lane & 15)) * lda + khalf;
  const float* ap3 = A + (size_t)(m0 + 48 + (lane & 15)) * lda + khalf;

  v8f acc0 = {}, acc1 = {}, acc2 = {}, acc3 = {};
  int Kfast = K & ~31;
  for (int k0 = 0; k0 < Kfast; k0 += 32) {
    v16bf b  = load_b_fast(W, Nc, k0, ncl, lane);
    v16bf a0 = load_a_vec(ap0 + k0);
    v16bf a1 = load_a_vec(ap1 + k0);
    v16bf a2 = load_a_vec(ap2 + k0);
    v16bf a3 = load_a_vec(ap3 + k0);
    acc0 = WMMA_BF16(a0, b, acc0);
    acc1 = WMMA_BF16(a1, b, acc1);
    acc2 = WMMA_BF16(a2, b, acc2);
    acc3 = WMMA_BF16(a3, b, acc3);
  }
  if (Kfast < K) {                          // uniform tail chunk
    int k0 = Kfast;
    v16bf b  = load_b_tail(W, K, Nc, k0, ncl, lane);
    v16bf a0 = load_a_tail(A, lda, K, m0 +  0, k0, lane);
    v16bf a1 = load_a_tail(A, lda, K, m0 + 16, k0, lane);
    v16bf a2 = load_a_tail(A, lda, K, m0 + 32, k0, lane);
    v16bf a3 = load_a_tail(A, lda, K, m0 + 48, k0, lane);
    acc0 = WMMA_BF16(a0, b, acc0);
    acc1 = WMMA_BF16(a1, b, acc1);
    acc2 = WMMA_BF16(a2, b, acc2);
    acc3 = WMMA_BF16(a3, b, acc3);
  }
  if (n < Nc) {
    float bv = bias[n];
    int rbase = m0 + ((lane & 16) ? 8 : 0);
#pragma unroll
    for (int v = 0; v < 8; ++v) {
      float x0 = acc0[v] + bv, x1 = acc1[v] + bv, x2 = acc2[v] + bv, x3 = acc3[v] + bv;
      if (relu) { x0 = fmaxf(x0, 0.f); x1 = fmaxf(x1, 0.f); x2 = fmaxf(x2, 0.f); x3 = fmaxf(x3, 0.f); }
      C[(size_t)(rbase + v +  0) * ldc + coff + n] = x0;
      C[(size_t)(rbase + v + 16) * ldc + coff + n] = x1;
      C[(size_t)(rbase + v + 32) * ldc + coff + n] = x2;
      C[(size_t)(rbase + v + 48) * ldc + coff + n] = x3;
    }
  }
}

// ---------------------------------------------------------------------------
// duplicate_up GEMM: virtual A row r = [FEAT[b, r%N] (360) | grid[(r/N)%4] (2)]
// K = 362 -> 11 fast chunks + 1 tail chunk; 64x16 per wave (rep uniform since
// 64 | 4096 -> a 64-row block never crosses a replica boundary).
// ---------------------------------------------------------------------------
__global__ __launch_bounds__(32, 1)
void gemm_dup(const float* __restrict__ FEAT,
              const float* __restrict__ W,   // 362 x 256
              const float* __restrict__ bias,
              float* __restrict__ C) {
  int lane = threadIdx.x;
  int m0 = blockIdx.x * 64;
  int n0 = blockIdx.y * 16;
  int n = n0 + (lane & 15);               // Nc = 256, always in range
  __builtin_prefetch(W, 0, 3);

  int rep = (m0 >> 12) & 3;               // uniform across the 64-row block
  float gx = c_grid[rep * 2 + 0], gy = c_grid[rep * 2 + 1];
  int khalf = (lane & 16) ? 8 : 0;
  const float* rowp[4];
#pragma unroll
  for (int t = 0; t < 4; ++t) {
    int r  = m0 + t * 16 + (lane & 15);
    int bb = r >> 14, pn = r & (N_ - 1);
    rowp[t] = FEAT + (size_t)(bb * N_ + pn) * 360;
  }

  v8f acc[4] = {{}, {}, {}, {}};
#pragma unroll
  for (int kc = 0; kc < 11; ++kc) {       // k0 = 0..320, all chunks < 360-31
    int k0 = kc * 32;
    v16bf b = load_b_fast(W, 256, k0, n, lane);
#pragma unroll
    for (int t = 0; t < 4; ++t) {
      v16bf a = load_a_vec(rowp[t] + k0 + khalf);
      acc[t] = WMMA_BF16(a, b, acc[t]);
    }
  }
  { // tail chunk: k = 352..383 (FEAT cols 352..359, grid 360..361, zero pad)
    int kb = 352 + khalf;
    v16bf b = load_b_tail(W, 362, 256, 352, n, lane);
#pragma unroll
    for (int t = 0; t < 4; ++t) {
      v16bf a;
#pragma unroll
      for (int jj = 0; jj < 16; ++jj) {
        int k = kb + ((jj < 8) ? jj : 8 + jj);   // elems 0..7 -> K+0..7, 8..15 -> K+16..23
        float v = rowp[t][k < 360 ? k : 359];
        v = (k < 360) ? v : ((k == 360) ? gx : ((k == 361) ? gy : 0.0f));
        a[jj] = f2bf(v);
      }
      acc[t] = WMMA_BF16(a, b, acc[t]);
    }
  }
  float bv = bias[n];
  int rbase = m0 + ((lane & 16) ? 8 : 0);
#pragma unroll
  for (int t = 0; t < 4; ++t)
#pragma unroll
    for (int v = 0; v < 8; ++v)
      C[(size_t)(rbase + t * 16 + v) * 256 + n] = acc[t][v] + bv;
}

// ---------------------------------------------------------------------------
// Training-mode BatchNorm over all rows, one channel per block (in place).
// ---------------------------------------------------------------------------
__global__ __launch_bounds__(256, 1)
void bn_act(float* __restrict__ X, int Mrows, int Cc,
            const float* __restrict__ gamma, const float* __restrict__ beta,
            int relu) {
  int c = blockIdx.x;
  __shared__ float s_sum[256], s_sq[256];
  float s = 0.0f, q = 0.0f;
  for (int r = threadIdx.x; r < Mrows; r += blockDim.x) {
    float v = X[(size_t)r * Cc + c];
    s += v; q += v * v;
  }
  s_sum[threadIdx.x] = s; s_sq[threadIdx.x] = q;
  __syncthreads();
  for (int off = 128; off > 0; off >>= 1) {
    if (threadIdx.x < off) {
      s_sum[threadIdx.x] += s_sum[threadIdx.x + off];
      s_sq[threadIdx.x]  += s_sq[threadIdx.x + off];
    }
    __syncthreads();
  }
  float inv = 1.0f / (float)Mrows;
  float mu  = s_sum[0] * inv;
  float var = s_sq[0] * inv - mu * mu;
  float rs  = rsqrtf(var + 1e-5f);
  float g = gamma[c], be = beta[c];
  for (int r = threadIdx.x; r < Mrows; r += blockDim.x) {
    size_t i = (size_t)r * Cc + c;
    float v = (X[i] - mu) * rs * g + be;
    if (relu) v = fmaxf(v, 0.0f);
    X[i] = v;
  }
}

// ---------------------------------------------------------------------------
// Fused EdgeConv: one wave per point. LDS (bf16):
//   EF: 32 x KP1 = [xt | nf | nf-xt | 0-pad]        (rows 24..31 unused)
//   Y : 32 x 160 = [y3(0:24) | y2(24:48) | y1(48:72) | xt(72:72+D) | 0-pad]
// Pad columns are zeroed (0 x Inf -> NaN would poison sums); pad rows are
// not (their garbage only reaches D rows the K-max never reads).
// ---------------------------------------------------------------------------
template <int LD, int KCH, int KREAL>
__device__ __forceinline__ void wave_gemm24(const __bf16 (*Ain)[LD], int acol,
                                            const float* __restrict__ W,
                                            const float* __restrict__ bias,
                                            __bf16 (*Yout)[160], int ycol,
                                            bool relu, int lane) {
#pragma unroll
  for (int nt = 0; nt < 2; ++nt) {
    int n0 = nt * 16;
    int n  = n0 + (lane & 15);
    int ncl = n < 24 ? n : 23;
    v8f acc0 = {}, acc1 = {};
#pragma unroll
    for (int kc = 0; kc < KCH; ++kc) {
      v16bf b = (kc * 32 + 32 <= KREAL)
                    ? load_b_fast(W, 24, kc * 32, ncl, lane)
                    : load_b_tail(W, KREAL, 24, kc * 32, ncl, lane);
      v16bf a0 = load_a_lds<LD>(Ain,  0, acol + kc * 32, lane);
      v16bf a1 = load_a_lds<LD>(Ain, 16, acol + kc * 32, lane);
      acc0 = WMMA_BF16(a0, b, acc0);
      acc1 = WMMA_BF16(a1, b, acc1);
    }
    if (n < 24) {
      float bv = bias[n];
#pragma unroll
      for (int v = 0; v < 8; ++v) {
        int row = ((lane & 16) ? 8 : 0) + v;
        float x0 = acc0[v] + bv, x1 = acc1[v] + bv;
        if (relu) { x0 = fmaxf(x0, 0.0f); x1 = fmaxf(x1, 0.0f); }
        Yout[row][ycol + n]      = f2bf(x0);
        Yout[row + 16][ycol + n] = f2bf(x1);
      }
    }
  }
}

template <int D>
__global__ __launch_bounds__(64, 1)
void edgeconv_kernel(const float* __restrict__ CUR, int ldcur,
                     const int* __restrict__ IDX,
                     const float* __restrict__ w1, const float* __restrict__ b1,
                     const float* __restrict__ wm, const float* __restrict__ bm,
                     const float* __restrict__ wl, const float* __restrict__ bl,
                     float* __restrict__ FEAT, int ldf, int coff) {
  constexpr int C    = 24;
  constexpr int K1   = 3 * D;               // 72 / 144
  constexpr int KP1  = (K1 + 31) & ~31;     // 96 / 160
  constexpr int YW   = 160;
  constexpr int OUTW = 3 * C + D;           // 96 / 120
  constexpr int WAVES = 2;

  __shared__ __bf16 sEF[WAVES][32][KP1];
  __shared__ __bf16 sY[WAVES][32][YW];

  int wave = threadIdx.x >> 5;
  int lane = threadIdx.x & 31;
  int point = blockIdx.x * WAVES + wave;    // 0..BN_-1
  int b = point >> 12;

  __bf16 (*EF)[KP1] = sEF[wave];
  __bf16 (*Y)[YW]   = sY[wave];

  // --- build edge features: lane r owns EF/Y row r (rows 0..23) ------------
  if (lane < 24) {
    int r = lane;
    int nb = IDX[(size_t)point * KNN_K + r];
    const float* xt = CUR + (size_t)point * ldcur;          // 16B aligned
    const float* nf = CUR + (size_t)(b * N_ + nb) * ldcur;  // 16B aligned
#pragma unroll
    for (int k = 0; k < D; k += 8) {
      float4v xa = *(const float4v*)(xt + k);
      float4v xb = *(const float4v*)(xt + k + 4);
      float4v na = *(const float4v*)(nf + k);
      float4v nb4 = *(const float4v*)(nf + k + 4);
      v8bf xv = cvt8(xa, xb);
      v8bf nv = cvt8(na, nb4);
      v8bf dv = cvt8(na - xa, nb4 - xb);
      *(v8bf*)&EF[r][k]          = xv;
      *(v8bf*)&EF[r][D + k]      = nv;
      *(v8bf*)&EF[r][2 * D + k]  = dv;
      *(v8bf*)&Y[r][3 * C + k]   = xv;      // xt slot of concat buffer
    }
    v8bf z = {};
#pragma unroll
    for (int k = K1; k < KP1; k += 8)       *(v8bf*)&EF[r][k] = z;   // pad cols
#pragma unroll
    for (int k = 3 * C + D; k < YW; k += 8) *(v8bf*)&Y[r][k]  = z;   // pad cols
  }
  __builtin_amdgcn_wave_barrier();

  // layer_first: EF(3D) @ w1 -> relu -> y1 (cols 48..71)
  wave_gemm24<KP1, KP1 / 32, K1>(EF, 0, w1, b1, Y, 48, true, lane);
  __builtin_amdgcn_wave_barrier();
  // middle: [y1|xt] (C+D) @ wm -> relu -> y2 (cols 24..47)
  wave_gemm24<YW, (C + D + 31) / 32, C + D>(Y, 48, wm, bm, Y, 24, true, lane);
  __builtin_amdgcn_wave_barrier();
  // layer_last: [y2|y1|xt] (2C+D) @ wl -> y3 (cols 0..23)
  wave_gemm24<YW, (2 * C + D + 31) / 32, 2 * C + D>(Y, 24, wl, bl, Y, 0, false, lane);
  __builtin_amdgcn_wave_barrier();

  // max over the 24 neighbor rows; write into concatenated feature buffer
  for (int col = lane; col < OUTW; col += 32) {
    float mx = -3.4e38f;
#pragma unroll
    for (int r = 0; r < 24; ++r) mx = fmaxf(mx, bf2f(Y[r][col]));
    FEAT[(size_t)point * ldf + coff + col] = mx;
  }
}

// ---------------------------------------------------------------------------
extern "C" void kernel_launch(void* const* d_in, const int* in_sizes, int n_in,
                              void* d_out, int out_size, void* d_ws, size_t ws_size,
                              hipStream_t stream) {
  (void)in_sizes; (void)n_in; (void)out_size; (void)ws_size;

  // ---- inputs (setup_inputs dict order; params pytree: sorted dict keys) --
  const float* points  = (const float*)d_in[0];
  const float* dup_b1  = (const float*)d_in[1];
  const float* dup_b2  = (const float*)d_in[2];
  const float* dup_be1 = (const float*)d_in[3];
  const float* dup_be2 = (const float*)d_in[4];
  const float* dup_g1  = (const float*)d_in[5];
  const float* dup_g2  = (const float*)d_in[6];
  const float* dup_w1  = (const float*)d_in[7];
  const float* dup_w2  = (const float*)d_in[8];
  const float* cw0 = (const float*)d_in[9];   const float* cb0 = (const float*)d_in[10];
  const float* cw1 = (const float*)d_in[11];  const float* cb1 = (const float*)d_in[12];
  const float* cw2 = (const float*)d_in[13];  const float* cb2 = (const float*)d_in[14];
  // edge blocks: keys sorted -> b1, bl, bm, w1, wl, wm
  const float* e0_b1 = (const float*)d_in[15]; const float* e0_bl = (const float*)d_in[16];
  const float* e0_bm = (const float*)d_in[17]; const float* e0_w1 = (const float*)d_in[18];
  const float* e0_wl = (const float*)d_in[19]; const float* e0_wm = (const float*)d_in[20];
  const float* e1_b1 = (const float*)d_in[21]; const float* e1_bl = (const float*)d_in[22];
  const float* e1_bm = (const float*)d_in[23]; const float* e1_w1 = (const float*)d_in[24];
  const float* e1_wl = (const float*)d_in[25]; const float* e1_wm = (const float*)d_in[26];
  const float* e2_b1 = (const float*)d_in[27]; const float* e2_bl = (const float*)d_in[28];
  const float* e2_bm = (const float*)d_in[29]; const float* e2_w1 = (const float*)d_in[30];
  const float* e2_wl = (const float*)d_in[31]; const float* e2_wm = (const float*)d_in[32];
  const float* rb1 = (const float*)d_in[33]; const float* rb2 = (const float*)d_in[34];
  const float* rb3 = (const float*)d_in[35];
  const float* rw1 = (const float*)d_in[36]; const float* rw2 = (const float*)d_in[37];
  const float* rw3 = (const float*)d_in[38];

  // ---- workspace carving --------------------------------------------------
  char* ws = (char*)d_ws;
  size_t o = 0;
  auto carve = [&](size_t bytes) { char* p = ws + o; o = (o + bytes + 255) & ~(size_t)255; return p; };
  int*   IDX  = (int*)  carve((size_t)BN_ * KNN_K * sizeof(int));
  float* FEAT = (float*)carve((size_t)BN_ * 360 * sizeof(float));
  float* CUR  = (float*)carve((size_t)BN_ * 48 * sizeof(float));
  float* H1   = (float*)carve((size_t)M2_ * 256 * sizeof(float));  // reused for reg hidden 1
  float* H2   = (float*)carve((size_t)M2_ * 128 * sizeof(float));
  float* H4   = (float*)carve((size_t)M2_ * 64 * sizeof(float));
  float* OUT  = (float*)d_out;

  // ---- 1. KNN -------------------------------------------------------------
  knn_kernel<<<BN_ / 128, 128, 0, stream>>>(points, IDX);

  // ---- 2. GCN block 0: conv (3->24) into FEAT[:, :24] == cur0 -------------
  gemm_bias_act<<<dim3(BN_ / 64, 2), 32, 0, stream>>>(points, 3, 3, cw0, 24, cb0,
                                                      FEAT, 360, 0, 0);
  edgeconv_kernel<24><<<BN_ / 2, 64, 0, stream>>>(FEAT, 360, IDX,
                                                  e0_w1, e0_b1, e0_wm, e0_bm, e0_wl, e0_bl,
                                                  FEAT, 360, 24);
  // ---- 3. block 1: conv (120->48), edgeconv d=48 --------------------------
  gemm_bias_act<<<dim3(BN_ / 64, 3), 32, 0, stream>>>(FEAT, 360, 120, cw1, 48, cb1,
                                                      CUR, 48, 0, 0);
  edgeconv_kernel<48><<<BN_ / 2, 64, 0, stream>>>(CUR, 48, IDX,
                                                  e1_w1, e1_b1, e1_wm, e1_bm, e1_wl, e1_bl,
                                                  FEAT, 360, 120);
  // ---- 4. block 2: conv (240->48), edgeconv d=48 --------------------------
  gemm_bias_act<<<dim3(BN_ / 64, 3), 32, 0, stream>>>(FEAT, 360, 240, cw2, 48, cb2,
                                                      CUR, 48, 0, 0);
  edgeconv_kernel<48><<<BN_ / 2, 64, 0, stream>>>(CUR, 48, IDX,
                                                  e2_w1, e2_b1, e2_wm, e2_bm, e2_wl, e2_bl,
                                                  FEAT, 360, 240);

  // ---- 5. duplicate_up ----------------------------------------------------
  gemm_dup<<<dim3(M2_ / 64, 16), 32, 0, stream>>>(FEAT, dup_w1, dup_b1, H1);
  bn_act<<<256, 256, 0, stream>>>(H1, M2_, 256, dup_g1, dup_be1, 1);
  gemm_bias_act<<<dim3(M2_ / 64, 8), 32, 0, stream>>>(H1, 256, 256, dup_w2, 128, dup_b2,
                                                      H2, 128, 0, 0);
  bn_act<<<128, 256, 0, stream>>>(H2, M2_, 128, dup_g2, dup_be2, 0);

  // ---- 6. regressor -------------------------------------------------------
  gemm_bias_act<<<dim3(M2_ / 64, 16), 32, 0, stream>>>(H2, 128, 128, rw1, 256, rb1,
                                                       H1, 256, 0, 1);
  gemm_bias_act<<<dim3(M2_ / 64, 4), 32, 0, stream>>>(H1, 256, 256, rw2, 64, rb2,
                                                      H4, 64, 0, 1);
  gemm_bias_act<<<dim3(M2_ / 64, 1), 32, 0, stream>>>(H4, 64, 64, rw3, 3, rb3,
                                                      OUT, 3, 0, 0);
}
